// SVDMamba_37641093382204
// MI455X (gfx1250) — matrix operand order repaired and checked by reading, not compiled, not run on hardware
//
#include <hip/hip_runtime.h>

// ---------------------------------------------------------------------------
// Mamba block forward on gfx1250 (MI455X).
//   * All four matmuls via v_wmma_f32_16x16x32_bf16 (f32 accumulate).
//   * Selective scan as a chunked two-pass linear-recurrence scan:
//     exp-product identity  prod exp(dt*A) = exp(A*sum dt)  makes the chunk
//     combine exact and cheap.  64 chunks x 64 channel-waves = 4096 waves of
//     parallel work instead of 128 serial waves.
// ---------------------------------------------------------------------------

#define BB   2
#define LL   4096
#define DM   1024
#define DI   2048
#define DS   16
#define DTRK 64
#define ROWS (BB * LL)       // 8192 token rows
#define CL   128             // scan chunk length
#define NCB  (LL / CL)       // 32 chunks per batch
#define NCT  (ROWS / CL)     // 64 chunks total (never straddles batch)

typedef __attribute__((ext_vector_type(16))) __bf16 v16bf;
typedef __attribute__((ext_vector_type(8)))  __bf16 v8bf;
typedef __attribute__((ext_vector_type(8)))  float  v8f;

// ---------------------------------------------------------------- converts
__global__ __launch_bounds__(256)
void k_f32_to_bf16(const float* __restrict__ src, __bf16* __restrict__ dst, int n) {
    int i = blockIdx.x * blockDim.x + threadIdx.x;
    if (i < n) dst[i] = (__bf16)src[i];
}

__global__ __launch_bounds__(256)
void k_aneg(const float* __restrict__ A_log, float* __restrict__ Aneg, int n) {
    int i = blockIdx.x * blockDim.x + threadIdx.x;
    if (i < n) Aneg[i] = -__expf(A_log[i]);
}

// extract dt_in (first 64 cols of x_dbl [ROWS,96]) as bf16 [ROWS,64]
__global__ __launch_bounds__(256)
void k_extract_dtin(const float* __restrict__ x_dbl, __bf16* __restrict__ dtin) {
    int i = blockIdx.x * blockDim.x + threadIdx.x;
    if (i >= ROWS * DTRK) return;
    int r = i / DTRK, c = i - r * DTRK;
    dtin[i] = (__bf16)x_dbl[r * 96 + c];
}

// ------------------------------------------------------------- WMMA GEMM
// C[M,N] (f32) = A[M,K] (bf16 row-major) x W[N,K]^T (bf16 row-major).
// One wave -> 16x(16*TN) tile; 4 waves per block stacked along M.
template <int TN>
__global__ __launch_bounds__(128)
void k_gemm_bf16_wmma(const __bf16* __restrict__ A, const __bf16* __restrict__ W,
                      float* __restrict__ C, int M, int N, int K) {
    const int lane = threadIdx.x & 31;
    const int wave = threadIdx.x >> 5;
    const int m0   = (blockIdx.y * 4 + wave) * 16;   // wave-uniform
    if (m0 >= M) return;
    const int nlo  = lane & 15;
    const int sel  = lane >> 4;

    v8f acc[TN];
#pragma unroll
    for (int t = 0; t < TN; ++t)
#pragma unroll
        for (int r = 0; r < 8; ++r) acc[t][r] = 0.0f;

    const __bf16* arow = A + (size_t)(m0 + nlo) * K;
    const __bf16* wrow[TN];
#pragma unroll
    for (int t = 0; t < TN; ++t)
        wrow[t] = W + (size_t)((blockIdx.x * TN + t) * 16 + nlo) * K + sel * 16;

    for (int k = 0; k < K; k += 32) {
        const v8bf a0 = *(const v8bf*)(arow + k + sel * 16);
        const v8bf a1 = *(const v8bf*)(arow + k + sel * 16 + 8);
        v16bf afrag;
#pragma unroll
        for (int i = 0; i < 8; ++i) { afrag[i] = a0[i]; afrag[8 + i] = a1[i]; }
#pragma unroll
        for (int t = 0; t < TN; ++t) {
            const v16bf bfrag = *(const v16bf*)(wrow[t] + k);
            acc[t] = __builtin_amdgcn_wmma_f32_16x16x32_bf16(
                false, afrag, false, bfrag, (short)0, acc[t], false, false);
        }
    }
#pragma unroll
    for (int t = 0; t < TN; ++t) {
        const int n = (blockIdx.x * TN + t) * 16 + nlo;
#pragma unroll
        for (int r = 0; r < 8; ++r)
            C[(size_t)(m0 + sel * 8 + r) * N + n] = acc[t][r];
    }
}

// ------------------------------------------- causal depthwise conv + SiLU
__global__ __launch_bounds__(256)
void k_conv_silu(const float* __restrict__ xz, const float* __restrict__ cw,
                 const float* __restrict__ cb, __bf16* __restrict__ ucb) {
    int idx = blockIdx.x * blockDim.x + threadIdx.x;   // over ROWS*DI
    if (idx >= ROWS * DI) return;
    const int d  = idx & (DI - 1);
    const int bl = idx >> 11;
    const int l  = bl & (LL - 1);
    float acc = cb[d];
#pragma unroll
    for (int j = 0; j < 4; ++j) {
        const int ls = l - 3 + j;
        if (ls >= 0)
            acc += xz[(size_t)(bl - 3 + j) * (2 * DI) + d] * cw[d * 4 + j];
    }
    ucb[idx] = (__bf16)(acc / (1.0f + __expf(-acc)));  // SiLU
}

// --------------------------------------------------- scan pass 1 (chunks)
// One wave32 block = one chunk x 32 channels.  Local recurrence from h=0,
// emits local y, within-chunk cumulative dt (in-place over dtr), chunk-end
// state E and chunk dt-sum S.  B_t/C_t broadcast via wave32 shuffles.
__global__ __launch_bounds__(32)
void k_scan_pass1(const float* __restrict__ x_dbl, float* __restrict__ dtr,
                  const float* __restrict__ b_dt, const float* __restrict__ Aneg,
                  const __bf16* __restrict__ ucb, float* __restrict__ yloc,
                  float* __restrict__ Echk, float* __restrict__ Schk) {
    const int lane = threadIdx.x;
    const int c    = blockIdx.x >> 6;          // chunk 0..NCT-1   (DI/32 == 64)
    const int d    = (blockIdx.x & 63) * 32 + lane;

    float Ad[DS], h[DS];
#pragma unroll
    for (int s = 0; s < DS; ++s) { Ad[s] = Aneg[d * DS + s]; h[s] = 0.0f; }
    const float bdt = b_dt[d];
    float cds = 0.0f;

    for (int i = 0; i < CL; ++i) {
        const size_t row = (size_t)c * CL + i;
        const float vBC  = x_dbl[row * 96 + 64 + lane];         // B_t|C_t
        const float dx   = dtr[row * DI + d] + bdt;
        const float dt   = (dx > 20.0f) ? dx : __logf(1.0f + __expf(dx));
        cds += dt;
        dtr[row * DI + d] = cds;                                // cum-dt (in chunk)
        const float u   = (float)ucb[row * DI + d];
        const float dtu = dt * u;
        float y = 0.0f;
#pragma unroll
        for (int s = 0; s < DS; ++s) {
            const float Bs = __shfl(vBC, s, 32);
            const float Cs = __shfl(vBC, 16 + s, 32);
            const float dA = __expf(dt * Ad[s]);
            h[s] = fmaf(dA, h[s], dtu * Bs);
            y    = fmaf(h[s], Cs, y);
        }
        yloc[row * DI + d] = y;
    }
#pragma unroll
    for (int s = 0; s < DS; ++s) Echk[((size_t)c * DI + d) * DS + s] = h[s];
    Schk[(size_t)c * DI + d] = cds;
}

// ------------------------------------------------- scan fixup (tiny serial)
// H_{c+1} = exp(A*S_c) * H_c + E_c, reset at batch boundary.  One thread per
// (channel, state): 32K threads x 64 steps.
__global__ __launch_bounds__(256)
void k_scan_fixup(const float* __restrict__ Aneg, const float* __restrict__ Schk,
                  const float* __restrict__ Echk, float* __restrict__ Hchk) {
    const int i = blockIdx.x * blockDim.x + threadIdx.x;   // over DI*DS
    if (i >= DI * DS) return;
    const int d = i / DS, s = i - d * DS;
    const float Ads = Aneg[i];
    float h = 0.0f;
    for (int c = 0; c < NCT; ++c) {
        if ((c & (NCB - 1)) == 0) h = 0.0f;                // new batch
        const size_t cd = (size_t)c * DI + d;
        Hchk[cd * DS + s] = h;
        h = fmaf(__expf(Ads * Schk[cd]), h, Echk[cd * DS + s]);
    }
}

// ------------------------------------------- scan pass 2 (parallel fixup)
// y_true = y_local + sum_s exp(A_s*cumdt) * H_c[s] * C_l[s]; fused with the
// D*u skip and SiLU(z) gate; emits bf16 y for the out-proj WMMA GEMM.
__global__ __launch_bounds__(256)
void k_scan_pass2(const float* __restrict__ x_dbl, const float* __restrict__ cd,
                  const float* __restrict__ Aneg, const float* __restrict__ Hchk,
                  const float* __restrict__ yloc, const __bf16* __restrict__ ucb,
                  const float* __restrict__ xz, const float* __restrict__ Dsk,
                  __bf16* __restrict__ yb) {
    const int idx = blockIdx.x * blockDim.x + threadIdx.x;  // over ROWS*DI
    if (idx >= ROWS * DI) return;
    const int d   = idx & (DI - 1);
    const int row = idx >> 11;
    const int c   = row / CL;
    const float cds = cd[idx];
    const float* Hp = Hchk + ((size_t)c * DI + d) * DS;
    const float* Cp = x_dbl + (size_t)row * 96 + 80;
    const float* Ap = Aneg + d * DS;
    float corr = 0.0f;
#pragma unroll
    for (int s = 0; s < DS; ++s)
        corr = fmaf(__expf(Ap[s] * cds) * Hp[s], Cp[s], corr);
    const float u = (float)ucb[idx];
    const float z = xz[(size_t)row * (2 * DI) + DI + d];
    const float y = yloc[idx] + corr;
    yb[idx] = (__bf16)((y + u * Dsk[d]) * (z / (1.0f + __expf(-z))));
}

// ---------------------------------------------------------------------------
extern "C" void kernel_launch(void* const* d_in, const int* in_sizes, int n_in,
                              void* d_out, int out_size, void* d_ws, size_t ws_size,
                              hipStream_t stream) {
    const float* x       = (const float*)d_in[0];
    const float* W_in    = (const float*)d_in[1];
    const float* conv_w  = (const float*)d_in[2];
    const float* conv_b  = (const float*)d_in[3];
    const float* W_xproj = (const float*)d_in[4];
    const float* W_dt    = (const float*)d_in[5];
    const float* b_dt    = (const float*)d_in[6];
    const float* A_log   = (const float*)d_in[7];
    const float* Dv      = (const float*)d_in[8];
    const float* W_out   = (const float*)d_in[9];
    float* out = (float*)d_out;
    (void)in_sizes; (void)n_in; (void)out_size; (void)ws_size;

    char*  base = (char*)d_ws;
    size_t off  = 0;
    auto alloc = [&](size_t bytes) -> void* {
        void* r = base + off;
        off += (bytes + 255) & ~(size_t)255;
        return r;
    };
    __bf16* Xb     = (__bf16*)alloc((size_t)ROWS * DM   * 2);
    __bf16* Wb_in  = (__bf16*)alloc((size_t)2*DI * DM   * 2);
    float*  xz     = (float* )alloc((size_t)ROWS * 2*DI * 4);
    __bf16* ucb    = (__bf16*)alloc((size_t)ROWS * DI   * 2);
    __bf16* Wb_xp  = (__bf16*)alloc((size_t)96   * DI   * 2);
    float*  x_dbl  = (float* )alloc((size_t)ROWS * 96   * 4);
    __bf16* dtin   = (__bf16*)alloc((size_t)ROWS * DTRK * 2);
    __bf16* Wb_dt  = (__bf16*)alloc((size_t)DI   * DTRK * 2);
    float*  dtr    = (float* )alloc((size_t)ROWS * DI   * 4);  // dt -> cum-dt
    float*  yloc   = (float* )alloc((size_t)ROWS * DI   * 4);
    __bf16* yb     = (__bf16*)alloc((size_t)ROWS * DI   * 2);
    __bf16* Wb_out = (__bf16*)alloc((size_t)DM   * DI   * 2);
    float*  Aneg   = (float* )alloc((size_t)DI   * DS   * 4);
    float*  Echk   = (float* )alloc((size_t)NCT  * DI * DS * 4);
    float*  Schk   = (float* )alloc((size_t)NCT  * DI   * 4);
    float*  Hchk   = (float* )alloc((size_t)NCT  * DI * DS * 4);

    auto cvt = [&](const float* s, __bf16* dst, int n) {
        k_f32_to_bf16<<<(n + 255) / 256, 256, 0, stream>>>(s, dst, n);
    };
    cvt(x,       Xb,     ROWS * DM);
    cvt(W_in,    Wb_in,  2 * DI * DM);
    cvt(W_xproj, Wb_xp,  96 * DI);
    cvt(W_dt,    Wb_dt,  DI * DTRK);
    cvt(W_out,   Wb_out, DM * DI);
    k_aneg<<<(DI * DS + 255) / 256, 256, 0, stream>>>(A_log, Aneg, DI * DS);

    // in_proj: xz[8192,4096] = Xb x Wb_in^T
    k_gemm_bf16_wmma<4><<<dim3((2 * DI) / 64, ROWS / 64), 128, 0, stream>>>(
        Xb, Wb_in, xz, ROWS, 2 * DI, DM);

    // causal depthwise conv + SiLU -> ucb (bf16)
    k_conv_silu<<<(ROWS * DI + 255) / 256, 256, 0, stream>>>(xz, conv_w, conv_b, ucb);

    // x_proj: x_dbl[8192,96] = ucb x Wb_xp^T
    k_gemm_bf16_wmma<2><<<dim3(96 / 32, ROWS / 64), 128, 0, stream>>>(
        ucb, Wb_xp, x_dbl, ROWS, 96, DI);

    // dt_proj: dtr[8192,2048] = dtin x Wb_dt^T
    k_extract_dtin<<<(ROWS * DTRK + 255) / 256, 256, 0, stream>>>(x_dbl, dtin);
    k_gemm_bf16_wmma<4><<<dim3(DI / 64, ROWS / 64), 128, 0, stream>>>(
        dtin, Wb_dt, dtr, ROWS, DI, DTRK);

    // chunked selective scan: pass1 (4096 waves) -> fixup -> pass2 (parallel)
    k_scan_pass1<<<NCT * (DI / 32), 32, 0, stream>>>(
        x_dbl, dtr, b_dt, Aneg, ucb, yloc, Echk, Schk);
    k_scan_fixup<<<(DI * DS + 255) / 256, 256, 0, stream>>>(Aneg, Schk, Echk, Hchk);
    k_scan_pass2<<<(ROWS * DI + 255) / 256, 256, 0, stream>>>(
        x_dbl, dtr, Aneg, Hchk, yloc, ucb, xz, Dv, yb);

    // out_proj: out[8192,1024] = yb x Wb_out^T
    k_gemm_bf16_wmma<4><<<dim3(DM / 64, ROWS / 64), 128, 0, stream>>>(
        yb, Wb_out, out, ROWS, DM, DI);
}
